// MultiHeadLatentAttention_82274393522642
// MI455X (gfx1250) — matrix operand order repaired and checked by reading, not compiled, not run on hardware
//
#include <hip/hip_runtime.h>
#include <hip/hip_bf16.h>

typedef __attribute__((ext_vector_type(16))) _Float16 v16h;
typedef __attribute__((ext_vector_type(8)))  _Float16 v8h;
typedef __attribute__((ext_vector_type(8)))  float    v8f;
typedef __attribute__((ext_vector_type(4)))  unsigned int v4u;
typedef __attribute__((ext_vector_type(8)))  int      v8i;
typedef __attribute__((ext_vector_type(4)))  int      v4i;

#define NHEADS 16
#define HDIM   128
#define SEQ    2048
#define BATCH  2
#define HTOT   2048
#define LATD   512
#define MROWS  4096   /* B*S */

#ifndef __has_builtin
#define __has_builtin(x) 0
#endif
#if __has_builtin(__builtin_amdgcn_tensor_load_to_lds) && \
    __has_builtin(__builtin_amdgcn_s_wait_tensorcnt)
#define HAVE_TDM 1
#else
#define HAVE_TDM 0
#endif
#if defined(__has_include)
#if __has_include(<hip/amd_detail/amd_gfx1250_TDM.h>)
#define TDM_6ARG 1
#else
#define TDM_6ARG 0
#endif
#else
#define TDM_6ARG 0
#endif

// ---------------------------------------------------------------------------
// Fragment loaders following CDNA5 WMMA VGPR layouts (ISA 7.12.2).
// A (16x32 f16): lane L<16 -> row M=L, K in {0..7,16..23}; L>=16 -> row L-16,
//                K in {8..15,24..31}  => two 16B loads per lane.
// B (32x16 f16) for C = A * W^T with W row-major [N x K]:
//   lane L -> col n = L%16, K in {0..7,16..23} (L<16) or {8..15,24..31}
//   => two 16B loads from a W row (same K order as A).
// ---------------------------------------------------------------------------
static __device__ __forceinline__ v16h frag_a(const _Float16* base, int ld,
                                              int row0, int k0, int lane) {
  int m   = row0 + (lane & 15);
  int sel = (lane & 16) ? 8 : 0;
  const _Float16* p = base + (size_t)m * ld + k0 + sel;
  v8h lo = *(const v8h*)(p);
  v8h hi = *(const v8h*)(p + 16);
  return __builtin_shufflevector(lo, hi, 0,1,2,3,4,5,6,7,8,9,10,11,12,13,14,15);
}

static __device__ __forceinline__ v16h frag_bT(const _Float16* W, int ldk,
                                               int n0, int k0, int lane) {
  int n  = n0 + (lane & 15);
  int kb = k0 + ((lane & 16) ? 8 : 0);
  const _Float16* p = W + (size_t)n * ldk + kb;
  v8h lo = *(const v8h*)(p);
  v8h hi = *(const v8h*)(p + 16);
  return __builtin_shufflevector(lo, hi, 0,1,2,3,4,5,6,7,8,9,10,11,12,13,14,15);
}

static __device__ __forceinline__ v8f wmma_f16(v16h a, v16h b, v8f c) {
  return __builtin_amdgcn_wmma_f32_16x16x32_f16(false, a, false, b,
                                                (short)0, c, false, false);
}

#if HAVE_TDM
// ---------------------------------------------------------------------------
// TDM: DMA a 2D tile (tile_w x tile_h, f16) from global into LDS.
// D# built per CDNA5 ISA.  pad fields add 8 halves of
// LDS padding per 128-half row (pad_interval=5 -> 64 DWORDs, pad_amount=3 ->
// 4 DWORDs) so LDS rows stride 136 halves (staggers the 64 banks).
// ---------------------------------------------------------------------------
#define TDM_PADBITS ((1u << 20) | (5u << 22) | (3u << 25))

static __device__ __forceinline__ void tdm_load_2d(
    unsigned lds_byte_off, const void* gptr,
    unsigned tile_w, unsigned tile_h,
    unsigned tensor_w, unsigned tensor_h,
    unsigned long long row_stride_elems, unsigned pad_bits) {
  unsigned long long ga = (unsigned long long)(uintptr_t)gptr;
  v4u g0;
  g0.x = 1u;                                    // count=1 user descriptor
  g0.y = lds_byte_off;                          // lds_addr (bytes)
  g0.z = (unsigned)ga;                          // global_addr[31:0]
  g0.w = (unsigned)((ga >> 32) & 0x1FFFFFFull)  // global_addr[56:32]
         | (2u << 30);                          // type=2 ("image")
  v8i g1;
  g1[0] = (int)((1u << 16) | pad_bits);         // data_size=1 (2B) + pad cfg
  g1[1] = (int)((tensor_w & 0xFFFFu) << 16);    // tensor_dim0[15:0]
  g1[2] = (int)((tensor_w >> 16) | ((tensor_h & 0xFFFFu) << 16));
  g1[3] = (int)(((tensor_h >> 16) & 0xFFFFu) | (tile_w << 16));  // tile_dim0
  g1[4] = (int)tile_h;                          // tile_dim1 (tile_dim2 = 0)
  g1[5] = (int)(row_stride_elems & 0xFFFFFFFFull);   // dim0_stride[31:0]
  g1[6] = (int)((row_stride_elems >> 32) & 0xFFFFull);
  g1[7] = 0;
  v4i z4 = {0, 0, 0, 0};
#if TDM_6ARG
  v8i z8 = {0, 0, 0, 0, 0, 0, 0, 0};
  __builtin_amdgcn_tensor_load_to_lds(g0, g1, z4, z4, z8, 0);
#else
  __builtin_amdgcn_tensor_load_to_lds(g0, g1, z4, z4, 0);
#endif
}
#endif  // HAVE_TDM

// ---------------------------------------------------------------------------
// f32 -> f16 cast
// ---------------------------------------------------------------------------
__global__ void cast_f16_kernel(const float* __restrict__ src,
                                _Float16* __restrict__ dst, int n) {
  int i = blockIdx.x * 256 + threadIdx.x;
  if (i < n) dst[i] = (_Float16)src[i];
}

// ---------------------------------------------------------------------------
// Generic GEMM:  C[M x N] = A[M x K] * W[N x K]^T + bias[N]
// WG = 4 waves (128 thr), WG tile 128x128, wave tile 64x64 (16 WMMA accums).
// A and W stream from L2 (weights+activations fit in the 192 MB L2).
// ---------------------------------------------------------------------------
__global__ __launch_bounds__(128) void gemm_wmma_kernel(
    const _Float16* __restrict__ A, const _Float16* __restrict__ W,
    const float* __restrict__ bias, void* __restrict__ Cout,
    int M, int N, int K, int out_f32) {
  int lane = threadIdx.x & 31;
  int wave = threadIdx.x >> 5;
  int rowBase = blockIdx.y * 128 + (wave >> 1) * 64;
  int colBase = blockIdx.x * 128 + (wave & 1) * 64;

  v8f acc[4][4] = {};
  for (int k0 = 0; k0 < K; k0 += 32) {
    v16h af[4], bf[4];
#pragma unroll
    for (int i = 0; i < 4; ++i) af[i] = frag_a(A, K, rowBase + 16 * i, k0, lane);
#pragma unroll
    for (int j = 0; j < 4; ++j) bf[j] = frag_bT(W, K, colBase + 16 * j, k0, lane);
#pragma unroll
    for (int i = 0; i < 4; ++i)
#pragma unroll
      for (int j = 0; j < 4; ++j)
        acc[i][j] = wmma_f16(af[i], bf[j], acc[i][j]);
  }

  int rofs  = (lane & 16) ? 8 : 0;
  int col_l = lane & 15;
#pragma unroll
  for (int j = 0; j < 4; ++j) {
    int col  = colBase + 16 * j + col_l;
    float bv = bias[col];
#pragma unroll
    for (int i = 0; i < 4; ++i) {
      int row = rowBase + 16 * i + rofs;
#pragma unroll
      for (int r = 0; r < 8; ++r) {
        float v = acc[i][j][r] + bv;
        size_t idx = (size_t)(row + r) * N + col;
        if (out_f32) ((float*)Cout)[idx] = v;
        else         ((_Float16*)Cout)[idx] = (_Float16)v;
      }
    }
  }
}

// ---------------------------------------------------------------------------
// Assemble q or k: half (B*S x 1024) -> out (B, NH, S, HD) with
// cols 0..1023 = half, cols 1024..2047 = rotary(half).
// ---------------------------------------------------------------------------
__global__ void assemble_qk_kernel(const _Float16* __restrict__ half_mat,
                                   _Float16* __restrict__ out) {
  size_t idx = (size_t)blockIdx.x * 256 + threadIdx.x;   // over B*NH*S*HD
  int d = (int)(idx & 127);
  size_t t = idx >> 7;
  int s = (int)(t & (SEQ - 1)); t >>= 11;
  int h = (int)(t & (NHEADS - 1));
  int b = (int)(t >> 4);
  int c = h * HDIM + d;
  const _Float16* row = half_mat + ((size_t)b * SEQ + s) * 1024;
  float val;
  if (c < 1024) {
    val = (float)row[c];
  } else {
    int cr = c - 1024;
    int j  = (cr < 512) ? cr : cr - 512;
    // inv_freq = 10000^(-(j%256)/256); ln(10000)/256 = 0.0359778909f
    float ang = (float)s * __expf(-(float)(j & 255) * 0.0359778909f);
    float sn, cs;
    __sincosf(ang, &sn, &cs);
    float x1 = (float)row[j], x2 = (float)row[j + 512];
    val = (cr < 512) ? (x1 * cs - x2 * sn) : (x1 * sn + x2 * cs);
  }
  out[idx] = (_Float16)val;
}

// ---------------------------------------------------------------------------
// Assemble V transposed: vfull (B*S x 2048) -> vt (B, NH, HD, S)
// so PV B-fragments read contiguous K-runs along S.
// ---------------------------------------------------------------------------
__global__ void assemble_v_kernel(const _Float16* __restrict__ vfull,
                                  _Float16* __restrict__ vt) {
  size_t idx = (size_t)blockIdx.x * 256 + threadIdx.x;   // over B*NH*HD*S
  int s = (int)(idx & (SEQ - 1));
  size_t t = idx >> 11;
  int d = (int)(t & 127); t >>= 7;
  int h = (int)(t & (NHEADS - 1));
  int b = (int)(t >> 4);
  vt[idx] = vfull[((size_t)b * SEQ + s) * HTOT + h * HDIM + d];
}

// ---------------------------------------------------------------------------
// Flash attention (no mask): per (b,h), online softmax over all 2048 keys.
// WG = 8 waves (256 thr) = 128 q rows; wave owns 16 rows.
// K/Vt tiles double-buffered in LDS via the Tensor Data Mover so the next
// block's DMA overlaps the current block's WMMA + softmax work.
// ---------------------------------------------------------------------------
__global__ __launch_bounds__(256) void attn_kernel(
    const _Float16* __restrict__ Q,   // (B,NH,S,HD)
    const _Float16* __restrict__ Km,  // (B,NH,S,HD)
    const _Float16* __restrict__ Vt,  // (B,NH,HD,S)
    _Float16* __restrict__ Out)       // (B,S,H)
{
#if HAVE_TDM
  __shared__ _Float16 Ktile[2][128][136];
  __shared__ _Float16 Vtile[2][128][136];
#endif
  __shared__ _Float16 Pl[8][16][136];
  const float scale = 0.08838834764831845f;  // 1/sqrt(128)

  int lane = threadIdx.x & 31;
  int wave = threadIdx.x >> 5;
  int bid  = blockIdx.x;
  int qblk = bid & 15;
  int h    = (bid >> 4) & (NHEADS - 1);
  int b    = bid >> 8;

  size_t bh = (size_t)b * NHEADS + h;
  const _Float16* Qb = Q  + bh * (size_t)SEQ * HDIM;
  const _Float16* Kb = Km + bh * (size_t)SEQ * HDIM;
  const _Float16* Vb = Vt + bh * (size_t)HDIM * SEQ;
  int qrow0 = qblk * 128 + wave * 16;

  // Q fragments are loop-invariant: preload all 4 K-steps.
  v16h qf[4];
#pragma unroll
  for (int ks = 0; ks < 4; ++ks) qf[ks] = frag_a(Qb, HDIM, qrow0, ks * 32, lane);

  v8f o[8] = {};
  float mrow[8], lrow[8];
#pragma unroll
  for (int r = 0; r < 8; ++r) { mrow[r] = -1e30f; lrow[r] = 0.f; }

#if HAVE_TDM
  // prologue: kick off the first K/Vt tile pair into buffer 0
  if (threadIdx.x < 32) {   // wave 0 issues DMAs (TDM ignores EXEC)
    tdm_load_2d((unsigned)(uintptr_t)&Ktile[0][0][0], Kb,
                128, 128, HDIM, SEQ, HDIM, TDM_PADBITS);
    tdm_load_2d((unsigned)(uintptr_t)&Vtile[0][0][0], Vb,
                128, 128, SEQ, HDIM, SEQ, TDM_PADBITS);
  }
#endif

  for (int kb = 0; kb < SEQ; kb += 128) {
#if HAVE_TDM
    int cur = (kb >> 7) & 1;
    int nxt = cur ^ 1;
    __syncthreads();   // consumers of buffer 'nxt' (from kb-128) are done
    if (threadIdx.x < 32) {
      __builtin_amdgcn_s_wait_tensorcnt(0);   // current tiles have landed
      if (kb + 128 < SEQ) {                   // prefetch next block async
        tdm_load_2d((unsigned)(uintptr_t)&Ktile[nxt][0][0],
                    Kb + (size_t)(kb + 128) * HDIM,
                    128, 128, HDIM, SEQ, HDIM, TDM_PADBITS);
        tdm_load_2d((unsigned)(uintptr_t)&Vtile[nxt][0][0],
                    Vb + (kb + 128),
                    128, 128, SEQ, HDIM, SEQ, TDM_PADBITS);
      }
    }
    __syncthreads();   // current tiles visible to all waves
    const _Float16* Ksrc = &Ktile[cur][0][0]; const int kld = 136; const int kofs = 0;
    const _Float16* Vsrc = &Vtile[cur][0][0]; const int vld = 136; const int vofs = 0;
#else
    const _Float16* Ksrc = Kb;   const int kld = HDIM; const int kofs = kb;
    const _Float16* Vsrc = Vb;   const int vld = SEQ;  const int vofs = kb;
#endif

    // S = Q * K^T  (16 x 128 per wave); B-frag rows are local key indices
    v8f sacc[8] = {};
#pragma unroll
    for (int ks = 0; ks < 4; ++ks) {
#pragma unroll
      for (int j = 0; j < 8; ++j) {
        v16h bf = frag_bT(Ksrc + (size_t)kofs * kld, kld, 16 * j, ks * 32, lane);
        sacc[j] = wmma_f16(qf[ks], bf, sacc[j]);
      }
    }

    // online softmax update (row state per 16-lane half)
    float newm[8], alpha[8], rsum[8];
#pragma unroll
    for (int r = 0; r < 8; ++r) {
      float mx = -1e30f;
#pragma unroll
      for (int j = 0; j < 8; ++j) mx = fmaxf(mx, sacc[j][r]);
      mx *= scale;
#pragma unroll
      for (int off = 1; off < 16; off <<= 1)
        mx = fmaxf(mx, __shfl_xor(mx, off, 32));
      newm[r] = fmaxf(mrow[r], mx);
    }
#pragma unroll
    for (int j = 0; j < 8; ++j)
#pragma unroll
      for (int r = 0; r < 8; ++r)
        sacc[j][r] = __expf(sacc[j][r] * scale - newm[r]);
#pragma unroll
    for (int r = 0; r < 8; ++r) {
      float sm = 0.f;
#pragma unroll
      for (int j = 0; j < 8; ++j) sm += sacc[j][r];
#pragma unroll
      for (int off = 1; off < 16; off <<= 1) sm += __shfl_xor(sm, off, 32);
      rsum[r] = sm;
      alpha[r] = __expf(mrow[r] - newm[r]);
      lrow[r] = lrow[r] * alpha[r] + rsum[r];
      mrow[r] = newm[r];
    }
#pragma unroll
    for (int j = 0; j < 8; ++j)
#pragma unroll
      for (int r = 0; r < 8; ++r) o[j][r] *= alpha[r];

    // C-layout -> A-layout via LDS
    __syncthreads();
    int rofs  = (lane & 16) ? 8 : 0;
    int col_l = lane & 15;
#pragma unroll
    for (int j = 0; j < 8; ++j)
#pragma unroll
      for (int r = 0; r < 8; ++r)
        Pl[wave][rofs + r][j * 16 + col_l] = (_Float16)sacc[j][r];
    __syncthreads();

    // O += P * V   (B[kk][d] = Vt[d][kk])
#pragma unroll
    for (int ks = 0; ks < 4; ++ks) {
      v16h pf = frag_a(&Pl[wave][0][0], 136, 0, ks * 32, lane);
#pragma unroll
      for (int j = 0; j < 8; ++j) {
        v16h bf = frag_bT(Vsrc + vofs, vld, 16 * j, ks * 32, lane);
        o[j] = wmma_f16(pf, bf, o[j]);
      }
    }
  }

  // epilogue: out[b, s, h*128 + d]
  int rofs  = (lane & 16) ? 8 : 0;
  int col_l = lane & 15;
  float inv[8];
#pragma unroll
  for (int r = 0; r < 8; ++r) inv[r] = 1.0f / lrow[r];
#pragma unroll
  for (int j = 0; j < 8; ++j) {
    int d = j * 16 + col_l;
#pragma unroll
    for (int r = 0; r < 8; ++r) {
      int s = qrow0 + rofs + r;
      Out[((size_t)b * SEQ + s) * HTOT + h * HDIM + d] =
          (_Float16)(o[j][r] * inv[r]);
    }
  }
}

// ---------------------------------------------------------------------------
// Host launcher
// ---------------------------------------------------------------------------
extern "C" void kernel_launch(void* const* d_in, const int* in_sizes, int n_in,
                              void* d_out, int out_size, void* d_ws, size_t ws_size,
                              hipStream_t stream) {
  const float* hs       = (const float*)d_in[0];
  const float* Wq_down  = (const float*)d_in[1];
  const float* bq_down  = (const float*)d_in[2];
  const float* Wkv_down = (const float*)d_in[3];
  const float* bkv_down = (const float*)d_in[4];
  const float* Wq_up    = (const float*)d_in[5];
  const float* bq_up    = (const float*)d_in[6];
  const float* Wk_up    = (const float*)d_in[7];
  const float* bk_up    = (const float*)d_in[8];
  const float* Wv_up    = (const float*)d_in[9];
  const float* bv_up    = (const float*)d_in[10];
  const float* Wo       = (const float*)d_in[11];
  const float* bo       = (const float*)d_in[12];
  float* out = (float*)d_out;

  const size_t N_HS   = (size_t)MROWS * HTOT;
  const size_t N_WQD  = (size_t)LATD * HTOT;
  const size_t N_WQU  = (size_t)1024 * LATD;
  const size_t N_WVU  = (size_t)HTOT * LATD;
  const size_t N_WO   = (size_t)HTOT * HTOT;
  const size_t N_LAT  = (size_t)MROWS * LATD;
  const size_t N_HALF = (size_t)MROWS * 1024;
  const size_t N_FULL = (size_t)MROWS * HTOT;

  _Float16* p = (_Float16*)d_ws;
  _Float16* hs16    = p; p += N_HS;
  _Float16* wqd16   = p; p += N_WQD;
  _Float16* wkvd16  = p; p += N_WQD;
  _Float16* wqu16   = p; p += N_WQU;
  _Float16* wku16   = p; p += N_WQU;
  _Float16* wvu16   = p; p += N_WVU;
  _Float16* wo16    = p; p += N_WO;
  _Float16* qlat16  = p; p += N_LAT;
  _Float16* kvlat16 = p; p += N_LAT;
  _Float16* qhalf16 = p; p += N_HALF;
  _Float16* khalf16 = p; p += N_HALF;
  _Float16* vfull16 = p; p += N_FULL;
  _Float16* q16     = p; p += N_FULL;
  _Float16* k16     = p; p += N_FULL;
  _Float16* vt16    = p; p += N_FULL;
  _Float16* attn16  = p; p += N_FULL;
  size_t need = (size_t)((char*)p - (char*)d_ws);
  if (ws_size < need) return;   // workspace too small: deterministic no-op

  auto cast = [&](const float* s, _Float16* d, size_t n) {
    cast_f16_kernel<<<dim3((unsigned)((n + 255) / 256)), dim3(256), 0, stream>>>(s, d, (int)n);
  };
  cast(hs, hs16, N_HS);
  cast(Wq_down, wqd16, N_WQD);
  cast(Wkv_down, wkvd16, N_WQD);
  cast(Wq_up, wqu16, N_WQU);
  cast(Wk_up, wku16, N_WQU);
  cast(Wv_up, wvu16, N_WVU);
  cast(Wo, wo16, N_WO);

  auto gemm = [&](const _Float16* A, const _Float16* W, const float* bias,
                  void* C, int M, int N, int K, int f32o) {
    gemm_wmma_kernel<<<dim3(N / 128, M / 128), dim3(128), 0, stream>>>(
        A, W, bias, C, M, N, K, f32o);
  };

  gemm(hs16, wqd16,  bq_down,  qlat16,  MROWS, LATD, HTOT, 0);
  gemm(hs16, wkvd16, bkv_down, kvlat16, MROWS, LATD, HTOT, 0);
  gemm(qlat16,  wqu16, bq_up, qhalf16, MROWS, 1024, LATD, 0);
  gemm(kvlat16, wku16, bk_up, khalf16, MROWS, 1024, LATD, 0);
  gemm(kvlat16, wvu16, bv_up, vfull16, MROWS, HTOT, LATD, 0);

  unsigned nblk = (unsigned)(N_FULL / 256);
  assemble_qk_kernel<<<dim3(nblk), dim3(256), 0, stream>>>(qhalf16, q16);
  assemble_qk_kernel<<<dim3(nblk), dim3(256), 0, stream>>>(khalf16, k16);
  assemble_v_kernel<<<dim3(nblk), dim3(256), 0, stream>>>(vfull16, vt16);

  attn_kernel<<<dim3(BATCH * NHEADS * (SEQ / 128)), dim3(256), 0, stream>>>(
      q16, k16, vt16, attn16);

  gemm(attn16, wo16, bo, out, MROWS, HTOT, HTOT, 1);
}
